// RelationalAttention_55001351192725
// MI455X (gfx1250) — matrix operand order repaired and checked by reading, not compiled, not run on hardware
//
#include <hip/hip_runtime.h>
#include <hip/hip_bf16.h>

// ---------------------------------------------------------------------------
// Relational attention, MI455X (gfx1250, wave32, WMMA, pipelined async-LDS).
//   B=2, S=1024, D=1024, H=16, KD=64, R=16, RP=64, HD=64
// Pipeline:
//   1) f32 -> f16 converts (x, symbols, 6 weight mats)
//   2) WMMA GEMM projections: xq,xk,qr,kr (from x), sv (from symbols)
//   3) scores = xq @ xk^T * ATTN_SCALE   (batched b,h; upper-tri blocks skipped)
//   4) causal softmax (in place, f16 storage / f32 math)
//   5) attended_symbols = attn @ sv      (batched, f32 out, K truncated at diag)
//   6) W2 = attn @ kr                    (batched, f16 out, K truncated at diag)
//   7) ar[i,h,r] = REL_SCALE * <qr[i,r,:], W2[h,i,r,:]>; mix with Wr;
//      combined = attended_symbols + attended_relations (f16)
//   8) out = combined @ Wo (f32 to d_out)
// ---------------------------------------------------------------------------

#define B_  2
#define S_  1024
#define D_  1024
#define H_  16
#define KD_ 64
#define R_  16
#define RP_ 64
#define HD_ 64

#define USE_ASYNC_LDS 1

static constexpr float ATTN_SCALE = 0.125f;  // 1/sqrt(64)
static constexpr float REL_SCALE  = 0.125f;  // 1/sqrt(64)

typedef _Float16 half_t;
typedef __attribute__((ext_vector_type(16))) _Float16 v16h;
typedef __attribute__((ext_vector_type(8)))  _Float16 v8h;
typedef __attribute__((ext_vector_type(8)))  float    v8f;
typedef long long ll;

// ---------------------------------------------------------------------------
// async 16B global -> LDS copy (CDNA5 GLOBAL_LOAD_ASYNC_TO_LDS_B128).
// lds_off: byte offset of destination in LDS; gptr: per-lane global source.
// Tracked with ASYNCcnt; consumer waits with s_wait_asynccnt <= N.
// ---------------------------------------------------------------------------
__device__ __forceinline__ void async_b128_to_lds(unsigned lds_off, const half_t* gptr) {
#if USE_ASYNC_LDS
  unsigned long long ga = (unsigned long long)(const void*)gptr;
  asm volatile("global_load_async_to_lds_b128 %0, %1, off"
               :: "v"(lds_off), "v"(ga) : "memory");
#else
  (void)lds_off; (void)gptr;
#endif
}
template <int N>
__device__ __forceinline__ void wait_async_le() {
#if USE_ASYNC_LDS
  asm volatile("s_wait_asynccnt %0" :: "i"(N) : "memory");
#endif
}

// ---------------------------------------------------------------------------
// f32 -> f16 convert (grid-stride)
// ---------------------------------------------------------------------------
__global__ void cvt_f32_to_f16(const float* __restrict__ src,
                               half_t* __restrict__ dst, int n) {
  int i = blockIdx.x * blockDim.x + threadIdx.x;
  int stride = gridDim.x * blockDim.x;
  for (; i < n; i += stride) dst[i] = (half_t)src[i];
}

// ---------------------------------------------------------------------------
// Tiled WMMA GEMM: C[M,N] = alpha * A[M,K] @ op(B)
//   A: f16 row-major, lda.   op(B): TRANS_B ? B is N x K row-major : K x N.
//   128 threads = 4 waves; block tile 64x64; each wave 32x32 (2x2 WMMA accs).
//   K stepped by 32 through double-buffered LDS; contiguous paths use
//   GLOBAL_LOAD_ASYNC_TO_LDS_B128 pipelined one tile ahead: while tile k is
//   consumed by WMMAs, tile k+1's async loads are in flight; we only wait
//   ASYNCcnt down to the count just issued (tile k's loads complete, k+1 not).
//   Batched via blockIdx.z: off = (z/HB)*s0 + (z%HB)*s1 per matrix.
//   tri_skip: skip blocks fully above the causal diagonal (scores GEMM).
//   kcausal : A rows are causal (A[m][k]==0 for k>m) -> K loop ends at m0+64.
//   Requires M%64==0, N%64==0, K%32==0 (true for every use here).
// ---------------------------------------------------------------------------
template<bool TRANS_B, bool OUT_F16>
__global__ __launch_bounds__(128)
void gemm_wmma(const half_t* __restrict__ A, const half_t* __restrict__ Bm,
               void* __restrict__ Cv, int M, int N, int K,
               int lda, int ldb, int ldc, float alpha, int HB,
               int tri_skip, int kcausal,
               ll sA0, ll sA1, ll sB0, ll sB1, ll sC0, ll sC1) {
  __shared__ __align__(32) half_t Asl[2 * 64 * 32];  // [buf][m][k]
  __shared__ __align__(32) half_t Bsl[2 * 64 * 32];  // [buf][n][k] (K-contig)

  const int n0 = blockIdx.x * 64;
  const int m0 = blockIdx.y * 64;
  if (tri_skip && n0 > m0 + 63) return;  // fully-masked causal block

  const int z = blockIdx.z;
  const size_t offA = (size_t)(z / HB) * sA0 + (size_t)(z % HB) * sA1;
  const size_t offB = (size_t)(z / HB) * sB0 + (size_t)(z % HB) * sB1;
  const size_t offC = (size_t)(z / HB) * sC0 + (size_t)(z % HB) * sC1;

  const int tid = threadIdx.x;
  const int lane = tid & 31;
  const int w = tid >> 5;
  const int wm = (w & 1) * 32;   // wave m-offset in block tile
  const int wn = (w >> 1) * 32;  // wave n-offset in block tile
  const int Kend = kcausal ? (m0 + 64) : K;

  // per-wave async instructions issued per tile (compile-time)
  constexpr int ASYNC_PER_TILE = TRANS_B ? 4 : 2;

  // ---- stage one K-tile into LDS buffer `buf`
  auto stage = [&](int k0, int buf) {
    half_t* As = Asl + buf * (64 * 32);
    half_t* Bs = Bsl + buf * (64 * 32);
    // A tile 64x32: two 16B chunks per thread, async
#pragma unroll
    for (int c = 0; c < 2; ++c) {
      const int e = (tid + c * 128) * 8;
      const int r = e >> 5, col = e & 31;
      const half_t* src = &A[offA + (size_t)(m0 + r) * lda + (k0 + col)];
#if USE_ASYNC_LDS
      async_b128_to_lds((unsigned)(uintptr_t)&As[e], src);
#else
      *reinterpret_cast<v8h*>(&As[e]) = *reinterpret_cast<const v8h*>(src);
#endif
    }
    if constexpr (TRANS_B) {
#pragma unroll
      for (int c = 0; c < 2; ++c) {
        const int e = (tid + c * 128) * 8;
        const int n = e >> 5, kk = e & 31;
        const half_t* src = &Bm[offB + (size_t)(n0 + n) * ldb + (k0 + kk)];
#if USE_ASYNC_LDS
        async_b128_to_lds((unsigned)(uintptr_t)&Bs[e], src);
#else
        *reinterpret_cast<v8h*>(&Bs[e]) = *reinterpret_cast<const v8h*>(src);
#endif
      }
    } else {
      // B row-major K x N: coalesced b128 reads along n, transposed LDS scatter
#pragma unroll
      for (int c = 0; c < 2; ++c) {
        const int e = (tid + c * 128) * 8;  // [k][n] linear, 8-elem chunks
        const int k = e >> 6, n = e & 63;
        v8h d = *reinterpret_cast<const v8h*>(
            &Bm[offB + (size_t)(k0 + k) * ldb + (n0 + n)]);
#pragma unroll
        for (int j = 0; j < 8; ++j) Bs[(n + j) * 32 + k] = d[j];
      }
    }
  };

  v8f acc[2][2];
#pragma unroll
  for (int mi = 0; mi < 2; ++mi)
#pragma unroll
    for (int ni = 0; ni < 2; ++ni)
#pragma unroll
      for (int e = 0; e < 8; ++e) acc[mi][ni][e] = 0.0f;

  stage(0, 0);  // prologue: tile 0 in flight

  for (int k0 = 0; k0 < Kend; k0 += 32) {
    const int cur = (k0 >> 5) & 1;
    if (k0 + 32 < Kend) {
      __syncthreads();               // everyone done reading buf cur^1
      stage(k0 + 32, cur ^ 1);       // issue next tile (stays in flight)
      wait_async_le<ASYNC_PER_TILE>();  // tile k complete; k+1 outstanding
    } else {
      wait_async_le<0>();            // drain
    }
    __syncthreads();                 // publish tile k across waves

    const half_t* As = Asl + cur * (64 * 32);
    const half_t* Bs = Bsl + cur * (64 * 32);

    // ---- build fragments (wave32 layouts, 16-bit A 16x32 / B 32x16)
    v16h af[2], bfr[2];
#pragma unroll
    for (int mi = 0; mi < 2; ++mi) {
      const int row = wm + mi * 16 + (lane & 15);
      const int kk = (lane < 16) ? 0 : 8;  // lanes 0-15: K 0..7/16..23; 16-31: 8..15/24..31
      v8h lo = *reinterpret_cast<const v8h*>(&As[row * 32 + kk]);
      v8h hi = *reinterpret_cast<const v8h*>(&As[row * 32 + 16 + kk]);
#pragma unroll
      for (int e = 0; e < 8; ++e) { af[mi][e] = lo[e]; af[mi][e + 8] = hi[e]; }
    }
#pragma unroll
    for (int ni = 0; ni < 2; ++ni) {
      const int n = wn + ni * 16 + (lane & 15);
      const int kk = (lane < 16) ? 0 : 8;  // lanes 0-15: K 0..7/16..23; 16-31: 8..15/24..31
      v8h lo = *reinterpret_cast<const v8h*>(&Bs[n * 32 + kk]);
      v8h hi = *reinterpret_cast<const v8h*>(&Bs[n * 32 + 16 + kk]);
#pragma unroll
      for (int e = 0; e < 8; ++e) { bfr[ni][e] = lo[e]; bfr[ni][e + 8] = hi[e]; }
    }

#pragma unroll
    for (int mi = 0; mi < 2; ++mi)
#pragma unroll
      for (int ni = 0; ni < 2; ++ni)
        acc[mi][ni] = __builtin_amdgcn_wmma_f32_16x16x32_f16(
            false, af[mi], false, bfr[ni], (short)0, acc[mi][ni], false, false);
  }

  // ---- store C (C/D layout: lane n = lane%16, VGPR v -> row v (+8 hi lanes))
#pragma unroll
  for (int mi = 0; mi < 2; ++mi)
#pragma unroll
    for (int ni = 0; ni < 2; ++ni) {
      const int gm = m0 + wm + mi * 16 + ((lane >> 4) << 3);
      const int gn = n0 + wn + ni * 16 + (lane & 15);
#pragma unroll
      for (int v = 0; v < 8; ++v) {
        float val = acc[mi][ni][v] * alpha;
        size_t o = offC + (size_t)(gm + v) * ldc + gn;
        if constexpr (OUT_F16) reinterpret_cast<half_t*>(Cv)[o] = (half_t)val;
        else                   reinterpret_cast<float*>(Cv)[o] = val;
      }
    }
}

// ---------------------------------------------------------------------------
// wave32 + block reductions
// ---------------------------------------------------------------------------
__device__ inline float wave_red_max(float v) {
#pragma unroll
  for (int m = 16; m >= 1; m >>= 1) v = fmaxf(v, __shfl_xor(v, m, 32));
  return v;
}
__device__ inline float wave_red_sum(float v) {
#pragma unroll
  for (int m = 16; m >= 1; m >>= 1) v += __shfl_xor(v, m, 32);
  return v;
}
__device__ inline float block_red_max(float v) {
  __shared__ float t[8];
  __syncthreads();
  v = wave_red_max(v);
  if ((threadIdx.x & 31) == 0) t[threadIdx.x >> 5] = v;
  __syncthreads();
  float r = t[0];
#pragma unroll
  for (int q = 1; q < 8; ++q) r = fmaxf(r, t[q]);
  return r;
}
__device__ inline float block_red_sum(float v) {
  __shared__ float t[8];
  __syncthreads();
  v = wave_red_sum(v);
  if ((threadIdx.x & 31) == 0) t[threadIdx.x >> 5] = v;
  __syncthreads();
  float r = 0.f;
#pragma unroll
  for (int q = 0; q < 8; ++q) r += t[q];
  return r;
}

// ---------------------------------------------------------------------------
// Causal softmax over one score row (scores pre-scaled by ATTN_SCALE).
//   blockIdx.x = (b*H + h)*S + i ; f16 storage, f32 math; masked tail -> 0.
// ---------------------------------------------------------------------------
__global__ __launch_bounds__(256)
void softmax_causal(half_t* __restrict__ attn) {
  const int row = blockIdx.x;
  const int i = row & (S_ - 1);
  half_t* p = attn + (size_t)row * S_;

  float mx = -INFINITY;
  for (int j = threadIdx.x; j <= i; j += 256) mx = fmaxf(mx, (float)p[j]);
  mx = block_red_max(mx);

  float sum = 0.f;
  for (int j = threadIdx.x; j <= i; j += 256) sum += __expf((float)p[j] - mx);
  sum = block_red_sum(sum);
  const float inv = 1.0f / sum;

  for (int j = threadIdx.x; j < S_; j += 256)
    p[j] = (j <= i) ? (half_t)(__expf((float)p[j] - mx) * inv) : (half_t)0.0f;
}

// ---------------------------------------------------------------------------
// ar[b,i,h,r] = REL_SCALE * <qr[b,i,r,:], W2[b,h,i,r,:]>
// attended_rel[b,i,h,d] = sum_r ar * Wr[h,d,r]
// combined(f16) = attended_symbols(f32) + attended_rel
// ---------------------------------------------------------------------------
__global__ __launch_bounds__(64)
void ar_mix(const half_t* __restrict__ qr, const half_t* __restrict__ W2,
            const float* __restrict__ Wr, const float* __restrict__ attended,
            half_t* __restrict__ combined) {
  const int idx = blockIdx.x;            // b*H*S + h*S + i
  const int i = idx & (S_ - 1);
  const int h = (idx >> 10) & (H_ - 1);
  const int b = idx >> 14;

  __shared__ float ar_s[R_];
  const int t = threadIdx.x;
  const half_t* qrow = qr + ((size_t)(b * S_ + i)) * (R_ * RP_);
  const half_t* wrow = W2 + (((size_t)(b * H_ + h)) * S_ + i) * (R_ * RP_);

  if (t < R_) {
    float s = 0.f;
#pragma unroll 8
    for (int p = 0; p < RP_; ++p)
      s += (float)qrow[t * RP_ + p] * (float)wrow[t * RP_ + p];
    ar_s[t] = s * REL_SCALE;
  }
  __syncthreads();

  float rel = 0.f;  // t == d
#pragma unroll
  for (int r = 0; r < R_; ++r) rel += ar_s[r] * Wr[(h * HD_ + t) * R_ + r];

  const size_t o = ((size_t)(b * S_ + i)) * (H_ * HD_) + h * HD_ + t;
  combined[o] = (half_t)(attended[o] + rel);
}

// ---------------------------------------------------------------------------
// host launcher
// ---------------------------------------------------------------------------
extern "C" void kernel_launch(void* const* d_in, const int* in_sizes, int n_in,
                              void* d_out, int out_size, void* d_ws, size_t ws_size,
                              hipStream_t stream) {
  const float* x        = (const float*)d_in[0];
  const float* symbols  = (const float*)d_in[1];
  const float* Wq_attn  = (const float*)d_in[2];
  const float* Wk_attn  = (const float*)d_in[3];
  const float* Wq_rel   = (const float*)d_in[4];
  const float* Wk_rel   = (const float*)d_in[5];
  const float* Wr       = (const float*)d_in[6];
  const float* Wv       = (const float*)d_in[7];
  const float* Wo       = (const float*)d_in[8];

  char* ws = (char*)d_ws;
  size_t off = 0;
  auto alloc = [&](size_t bytes) -> void* {
    void* p = ws + off;
    off = (off + bytes + 255) & ~(size_t)255;
    return p;
  };

  const int BS = B_ * S_;             // 2048
  const int DM = D_;                  // 1024
  const int BH = B_ * H_;             // 32

  half_t* xh   = (half_t*)alloc((size_t)BS * DM * 2);
  half_t* sh   = (half_t*)alloc((size_t)BS * DM * 2);
  half_t* wqa  = (half_t*)alloc((size_t)DM * DM * 2);
  half_t* wka  = (half_t*)alloc((size_t)DM * DM * 2);
  half_t* wqr  = (half_t*)alloc((size_t)DM * DM * 2);
  half_t* wkr  = (half_t*)alloc((size_t)DM * DM * 2);
  half_t* wvh  = (half_t*)alloc((size_t)DM * DM * 2);
  half_t* woh  = (half_t*)alloc((size_t)DM * DM * 2);
  half_t* xqh  = (half_t*)alloc((size_t)BS * DM * 2);
  half_t* xkh  = (half_t*)alloc((size_t)BS * DM * 2);
  half_t* qrh  = (half_t*)alloc((size_t)BS * DM * 2);
  half_t* krh  = (half_t*)alloc((size_t)BS * DM * 2);
  half_t* svh  = (half_t*)alloc((size_t)BS * DM * 2);
  half_t* attn = (half_t*)alloc((size_t)BH * S_ * S_ * 2);   // 64 MB
  half_t* W2   = (half_t*)alloc((size_t)BH * S_ * DM * 2);   // 64 MB
  float*  attd = (float*) alloc((size_t)BS * DM * 4);        // attended_symbols
  half_t* comb = (half_t*)alloc((size_t)BS * DM * 2);
  (void)ws_size; (void)in_sizes; (void)n_in; (void)out_size;

  // 1) converts
  cvt_f32_to_f16<<<1024, 256, 0, stream>>>(x, xh, BS * DM);
  cvt_f32_to_f16<<<1024, 256, 0, stream>>>(symbols, sh, BS * DM);
  cvt_f32_to_f16<<<1024, 256, 0, stream>>>(Wq_attn, wqa, DM * DM);
  cvt_f32_to_f16<<<1024, 256, 0, stream>>>(Wk_attn, wka, DM * DM);
  cvt_f32_to_f16<<<1024, 256, 0, stream>>>(Wq_rel, wqr, DM * DM);
  cvt_f32_to_f16<<<1024, 256, 0, stream>>>(Wk_rel, wkr, DM * DM);
  cvt_f32_to_f16<<<1024, 256, 0, stream>>>(Wv, wvh, DM * DM);
  cvt_f32_to_f16<<<1024, 256, 0, stream>>>(Wo, woh, DM * DM);

  const dim3 blk(128);
  // 2) projections: [2048,1024] @ [1024,1024] -> f16
  {
    const dim3 g(DM / 64, BS / 64, 1);
    gemm_wmma<false, true><<<g, blk, 0, stream>>>(xh, wqa, xqh, BS, DM, DM, DM, DM, DM, 1.0f, 1, 0, 0, 0, 0, 0, 0, 0, 0);
    gemm_wmma<false, true><<<g, blk, 0, stream>>>(xh, wka, xkh, BS, DM, DM, DM, DM, DM, 1.0f, 1, 0, 0, 0, 0, 0, 0, 0, 0);
    gemm_wmma<false, true><<<g, blk, 0, stream>>>(xh, wqr, qrh, BS, DM, DM, DM, DM, DM, 1.0f, 1, 0, 0, 0, 0, 0, 0, 0, 0);
    gemm_wmma<false, true><<<g, blk, 0, stream>>>(xh, wkr, krh, BS, DM, DM, DM, DM, DM, 1.0f, 1, 0, 0, 0, 0, 0, 0, 0, 0);
    gemm_wmma<false, true><<<g, blk, 0, stream>>>(sh, wvh, svh, BS, DM, DM, DM, DM, DM, 1.0f, 1, 0, 0, 0, 0, 0, 0, 0, 0);
  }
  // 3) scores = xq @ xk^T * ATTN_SCALE, per (b,h): M=N=1024, K=64.
  //    Blocks fully above the diagonal are skipped (softmax masks them).
  {
    const dim3 g(S_ / 64, S_ / 64, BH);
    gemm_wmma<true, true><<<g, blk, 0, stream>>>(
        xqh, xkh, attn, S_, S_, KD_, DM, DM, S_, ATTN_SCALE, H_,
        /*tri_skip=*/1, /*kcausal=*/0,
        (ll)S_ * DM, (ll)KD_,
        (ll)S_ * DM, (ll)KD_,
        (ll)H_ * S_ * S_, (ll)S_ * S_);
  }
  // 4) causal softmax in place
  softmax_causal<<<BH * S_, 256, 0, stream>>>(attn);
  // 5) attended_symbols = attn @ sv : per (b,h) M=1024, N=64, K=1024 -> f32.
  //    attn is causal -> truncate K at the diagonal.
  {
    const dim3 g(HD_ / 64, S_ / 64, BH);
    gemm_wmma<false, false><<<g, blk, 0, stream>>>(
        attn, svh, attd, S_, HD_, S_, S_, DM, DM, 1.0f, H_,
        /*tri_skip=*/0, /*kcausal=*/1,
        (ll)H_ * S_ * S_, (ll)S_ * S_,
        (ll)S_ * DM, (ll)HD_,
        (ll)S_ * DM, (ll)HD_);
  }
  // 6) W2 = attn @ kr : per (b,h) M=1024, N=1024, K=1024 -> f16 (K truncated)
  {
    const dim3 g(DM / 64, S_ / 64, BH);
    gemm_wmma<false, true><<<g, blk, 0, stream>>>(
        attn, krh, W2, S_, DM, S_, S_, DM, DM, 1.0f, H_,
        /*tri_skip=*/0, /*kcausal=*/1,
        (ll)H_ * S_ * S_, (ll)S_ * S_,
        (ll)S_ * DM, (ll)0,
        (ll)H_ * S_ * DM, (ll)S_ * DM);
  }
  // 7) rel mix + combine
  ar_mix<<<B_ * H_ * S_, 64, 0, stream>>>(qrh, W2, Wr, attd, comb);
  // 8) out = combined @ Wo -> f32
  {
    const dim3 g(DM / 64, BS / 64, 1);
    gemm_wmma<false, false><<<g, blk, 0, stream>>>(
        comb, woh, (float*)d_out, BS, DM, DM, DM, DM, DM, 1.0f, 1,
        0, 0, 0, 0, 0, 0, 0, 0);
  }
}
